// MultiHeadAttentionQuantum_65481071403818
// MI455X (gfx1250) — matrix operand order repaired and checked by reading, not compiled, not run on hardware
//
#include <hip/hip_runtime.h>

// ---------------------------------------------------------------------------
// Quantum-MHA collapse:  Y = cos(x + theta) [head<->qubit permuted] @ W^T + b
//   M = B = 32768, N = E = 1024, K = E = 1024  (H=256, Q=4)
// Strategy: fp16 WMMA (16x16x32) with f32 accumulate; permutation folded
// into a one-off fp16 repack of W; A computed on the fly (cos) into LDS.
// ---------------------------------------------------------------------------

typedef _Float16 v16h __attribute__((ext_vector_type(16)));
typedef float    v8f  __attribute__((ext_vector_type(8)));

union Frag16 { v16h v; uint4 q[2]; };      // natural alignment 32 from v16h
union Half4  { _Float16 h[4]; uint2 u; };

#define BM  64      // rows per workgroup
#define RS  40      // padded LDS row stride (halves): 80B -> bank-conflict-free-ish
#define EN  1024    // embed dim (N == K == EN)

// --- Kernel 1: repack W (fp32, row-major ExE) into permuted fp16 Wp --------
// Wp[j, k] = (half) W[j, (k&3)*256 + (k>>2)]   (undoes the (B,Q,H) transpose)
__global__ __launch_bounds__(256)
void qmha_pack_w(const float* __restrict__ W, _Float16* __restrict__ Wp) {
    int i = blockIdx.x * 256 + threadIdx.x;   // 0 .. EN*EN-1
    int j = i >> 10;
    int k = i & (EN - 1);
    int src = (j << 10) + ((k & 3) << 8) + (k >> 2);
    Wp[i] = (_Float16)W[src];
}

// --- Kernel 2: fused cos + GEMM via V_WMMA_F32_16X16X32_F16 ----------------
__global__ __launch_bounds__(512)
void qmha_gemm(const float* __restrict__ x,
               const float* __restrict__ theta,   // flat EN, k = h*4+q order
               const _Float16* __restrict__ Wp,   // permuted fp16 W, row-major
               const float* __restrict__ bias,
               float* __restrict__ out) {
    __shared__ _Float16 As[BM * RS];

    const int t    = threadIdx.x;
    const int lane = t & 31;
    const int wave = t >> 5;            // 0..15 : each wave owns 64 columns
    const int hl   = lane & 15;         // 0..15
    const int hi   = lane >> 4;         // 0 | 1
    const int m0   = blockIdx.x * BM;   // global row base

    // LDS fill mapping: 512 threads x 4 halves = 64 x 32 tile
    const int fm = t >> 3;              // row   0..63
    const int fk = (t & 7) << 2;        // kcol  0,4,..,28

    v8f acc[4][4];
#pragma unroll
    for (int mt = 0; mt < 4; ++mt)
#pragma unroll
        for (int nt = 0; nt < 4; ++nt)
            acc[mt][nt] = (v8f)(0.0f);

    const int kb = hi << 3;             // A lane K base: 0 or 8

    for (int k0 = 0; k0 < EN; k0 += 32) {
        // ---- stage A tile: cos(x + theta) -> fp16 in LDS ----
        float4 xv = *(const float4*)(x + (size_t)(m0 + fm) * EN + k0 + fk);
        float4 tv = *(const float4*)(theta + k0 + fk);
        Half4 f;
        f.h[0] = (_Float16)__cosf(xv.x + tv.x);
        f.h[1] = (_Float16)__cosf(xv.y + tv.y);
        f.h[2] = (_Float16)__cosf(xv.z + tv.z);
        f.h[3] = (_Float16)__cosf(xv.w + tv.w);
        *(uint2*)&As[fm * RS + fk] = f.u;
        __syncthreads();

        // ---- A fragments (ISA 16-bit A 16x32 layout): two b128 per tile ----
        Frag16 a[4];
#pragma unroll
        for (int mt = 0; mt < 4; ++mt) {
            const _Float16* r = &As[(mt * 16 + hl) * RS];
            a[mt].q[0] = *(const uint4*)(r + kb);        // K = kb .. kb+7
            a[mt].q[1] = *(const uint4*)(r + kb + 16);   // K = kb+16 .. kb+23
        }

        // ---- B fragments (lane n = column n, K = kb..kb+7 and kb+16..kb+23) ----
#pragma unroll
        for (int nt = 0; nt < 4; ++nt) {
            const int col = wave * 64 + nt * 16 + hl;
            const _Float16* bp = Wp + (size_t)col * EN + k0 + (hi << 3);
            Frag16 bf;
            bf.q[0] = *(const uint4*)(bp);
            bf.q[1] = *(const uint4*)(bp + 16);
#pragma unroll
            for (int mt = 0; mt < 4; ++mt) {
                acc[mt][nt] = __builtin_amdgcn_wmma_f32_16x16x32_f16(
                    /*neg_a=*/false, a[mt].v,
                    /*neg_b=*/false, bf.v,
                    /*c_mod=*/(short)0, acc[mt][nt],
                    /*reuse_a=*/false, /*reuse_b=*/false);
            }
        }
        __syncthreads();
    }

    // ---- epilogue: C layout (VGPR r -> row r / r+8; lane%16 -> col), +bias ----
#pragma unroll
    for (int nt = 0; nt < 4; ++nt) {
        const int colg = wave * 64 + nt * 16 + hl;
        const float bv = bias[colg];
#pragma unroll
        for (int mt = 0; mt < 4; ++mt) {
            const int rbase = m0 + mt * 16 + hi * 8;
            float* o = out + (size_t)rbase * EN + colg;
            v8f c = acc[mt][nt];
#pragma unroll
            for (int r = 0; r < 8; ++r)
                o[(size_t)r * EN] = c[r] + bv;
        }
    }
}

extern "C" void kernel_launch(void* const* d_in, const int* in_sizes, int n_in,
                              void* d_out, int out_size, void* d_ws, size_t ws_size,
                              hipStream_t stream) {
    const float* x     = (const float*)d_in[0];   // (B,1,E) fp32
    const float* theta = (const float*)d_in[1];   // (H,Q)   fp32, flat == k order
    const float* W     = (const float*)d_in[2];   // (E,E)   fp32
    const float* bias  = (const float*)d_in[3];   // (E,)    fp32
    float* out = (float*)d_out;

    _Float16* Wp = (_Float16*)d_ws;               // 2 MB scratch

    const int Btot = in_sizes[0] / EN;            // 32768

    qmha_pack_w<<<(EN * EN) / 256, 256, 0, stream>>>(W, Wp);
    qmha_gemm<<<Btot / BM, 512, 0, stream>>>(x, theta, Wp, bias, out);
}
